// RetNetDecoder_80195629351208
// MI455X (gfx1250) — matrix operand order repaired and checked by reading, not compiled, not run on hardware
//
#include <hip/hip_runtime.h>
#include <stdint.h>
#include <math.h>

// ---------------------------------------------------------------------------
// RetNet decoder forward for MI455X (gfx1250, wave32, WMMA).
// All GEMMs: f16 inputs (A row-major [M,K], B pre-transposed [N,K]),
// f32 accumulation via v_wmma_f32_16x16x32_f16, async global->LDS staging.
// ---------------------------------------------------------------------------

typedef _Float16 h16_t;
typedef _Float16 v8h  __attribute__((ext_vector_type(8)));
typedef _Float16 v16h __attribute__((ext_vector_type(16)));
typedef float    v8f  __attribute__((ext_vector_type(8)));
typedef float    v4f  __attribute__((ext_vector_type(4)));

#define B_    4
#define S_    1024
#define HID_  1024
#define H_    8
#define HD_   128
#define FFN_  4096
#define OUT_  512
#define M_    (B_*S_)          // 4096 rows
#define EPS_  1e-5f

// ---- CDNA5 helpers ---------------------------------------------------------

// Generic pointer to LDS: low 32 bits are the LDS byte offset (aperture model).
__device__ __forceinline__ uint32_t ldsOff(const void* p) {
  return (uint32_t)(uintptr_t)p;
}

// Async copy 16B global -> LDS per lane (tracked by ASYNCcnt).
__device__ __forceinline__ void asyncLds128(uint32_t lds, const void* g) {
  asm volatile("global_load_async_to_lds_b128 %0, %1, off"
               :: "v"(lds), "v"(g) : "memory");
}
__device__ __forceinline__ void waitAsyncLe4() {
  asm volatile("s_wait_asynccnt 0x4" ::: "memory");
}
__device__ __forceinline__ void waitAsyncLe0() {
  asm volatile("s_wait_asynccnt 0x0" ::: "memory");
}

__device__ __forceinline__ v16h cat8(v8h lo, v8h hi) {
  return __builtin_shufflevector(lo, hi, 0,1,2,3,4,5,6,7,8,9,10,11,12,13,14,15);
}
__device__ __forceinline__ v8f wmma16(v16h a, v16h b, v8f c) {
  return __builtin_amdgcn_wmma_f32_16x16x32_f16(false, a, false, b,
                                                (short)0, c, false, false);
}

// ---- generic tiled WMMA GEMM ----------------------------------------------
// C[M,N] = epi(A[M,K] @ Bt[N,K]^T).  BM=BN=128, BK=32, 8 waves, wave=32x64.
// EPI: 0 none, 1 +bias, 2 +bias,gelu(exact), 3 +residual, 4 +bias+residual,
//      5 softplus(0.5*(acc+bias))+1e-3
template<int EPI>
__global__ __launch_bounds__(256)
void k_gemm(const h16_t* __restrict__ A, const h16_t* __restrict__ Bt,
            const float* __restrict__ bias, const float* __restrict__ R,
            float* __restrict__ Cf, h16_t* __restrict__ Ch,
            int M, int N, int K)
{
  __shared__ __align__(16) h16_t sA[2][128*32];
  __shared__ __align__(16) h16_t sB[2][128*32];
  const int tid    = threadIdx.x;
  const int wave   = tid >> 5, lane = tid & 31;
  const int lane16 = lane & 15, laneHi = lane >> 4;
  const int wm = wave & 3, wn = wave >> 2;
  const int m0 = blockIdx.y * 128, n0 = blockIdx.x * 128;
  const int nk = K >> 5;

  const v8f vz = {0.f,0.f,0.f,0.f,0.f,0.f,0.f,0.f};
  v8f acc[2][4];
#pragma unroll
  for (int i = 0; i < 2; ++i)
#pragma unroll
    for (int j = 0; j < 4; ++j) acc[i][j] = vz;

  auto issue = [&](int kt, int buf) {
    const int k0 = kt << 5;
#pragma unroll
    for (int i = 0; i < 2; ++i) {
      int c = tid + (i << 8);
      int row = c >> 2, co = (c & 3) << 3;
      asyncLds128(ldsOff(&sA[buf][row*32 + co]),
                  A + (size_t)(m0 + row) * K + k0 + co);
    }
#pragma unroll
    for (int i = 0; i < 2; ++i) {
      int c = tid + (i << 8);
      int row = c >> 2, co = (c & 3) << 3;
      asyncLds128(ldsOff(&sB[buf][row*32 + co]),
                  Bt + (size_t)(n0 + row) * K + k0 + co);
    }
  };

  issue(0, 0);
  for (int kt = 0; kt < nk; ++kt) {
    const int buf = kt & 1;
    if (kt + 1 < nk) { issue(kt + 1, buf ^ 1); waitAsyncLe4(); }
    else             { waitAsyncLe0(); }
    __syncthreads();

    const h16_t* pa = sA[buf];
    const h16_t* pb = sB[buf];
    // A fragment: lanes 0-15 K=0..7 (v0-3) & 16..23 (v4-7); lanes16-31: +8
    v16h af[2];
#pragma unroll
    for (int mt = 0; mt < 2; ++mt) {
      int row = wm*32 + mt*16 + lane16;
      int kh  = laneHi << 3;
      v8h lo = *(const v8h*)(pa + row*32 + kh);
      v8h hi = *(const v8h*)(pa + row*32 + 16 + kh);
      af[mt] = cat8(lo, hi);
    }
    // B fragment: lanes 0-15 K=0..7 (v0-3) & 16..23 (v4-7); lanes16-31: +8
    v16h bf[4];
#pragma unroll
    for (int nt = 0; nt < 4; ++nt) {
      int row = wn*64 + nt*16 + lane16;
      int kh  = laneHi << 3;
      v8h lo = *(const v8h*)(pb + row*32 + kh);
      v8h hi = *(const v8h*)(pb + row*32 + 16 + kh);
      bf[nt] = cat8(lo, hi);
    }
#pragma unroll
    for (int mt = 0; mt < 2; ++mt)
#pragma unroll
      for (int nt = 0; nt < 4; ++nt)
        acc[mt][nt] = wmma16(af[mt], bf[nt], acc[mt][nt]);
    __syncthreads();
  }

  // Epilogue. C layout: VGPR r -> M = r + 8*laneHi, N = lane16.
#pragma unroll
  for (int mt = 0; mt < 2; ++mt) {
#pragma unroll
    for (int nt = 0; nt < 4; ++nt) {
      int n = n0 + wn*64 + nt*16 + lane16;
      float bv = (EPI==1 || EPI==2 || EPI==4 || EPI==5) ? bias[n] : 0.f;
#pragma unroll
      for (int r = 0; r < 8; ++r) {
        int m = m0 + wm*32 + mt*16 + r + (laneHi << 3);
        float v = acc[mt][nt][r] + bv;
        if (EPI == 3 || EPI == 4) v += R[(size_t)m * N + n];
        if (EPI == 2) v = 0.5f * v * (1.f + erff(v * 0.70710678118654752f));
        if (EPI == 5) {
          float t = 0.5f * v;
          v = ((t > 20.f) ? t : log1pf(expf(t))) + 0.001f;
        }
        if (Cf) Cf[(size_t)m * N + n] = v;
        if (Ch) Ch[(size_t)m * N + n] = (h16_t)v;
      }
    }
  }
}

// ---- fused retention + group norm -----------------------------------------
// Q,K,V: f16 [B*S, HID] (head-major columns h*128+e), xPos already applied.
// Per block: (b, h, 128 rows), 8 waves x 16-row strips. Streams 32-key blocks:
// scores = Q Kt via WMMA, decay/causal mask analytically, scores@V via WMMA.
// Epilogue fuses per-head group norm (rows live in 16-lane halves on wave32).
__global__ __launch_bounds__(256)
void k_retention(const h16_t* __restrict__ Q, const h16_t* __restrict__ Kh,
                 const h16_t* __restrict__ V, const float* __restrict__ gn_g,
                 const float* __restrict__ gn_b, h16_t* __restrict__ Yout)
{
  __shared__ __align__(16) h16_t sV[HD_*32];     // V^T tile [e][key]
  __shared__ __align__(16) h16_t sS[8][16*32];   // per-wave score tile [m][key]
  const int tid    = threadIdx.x;
  const int wave   = tid >> 5, lane = tid & 31;
  const int lane16 = lane & 15, laneHi = lane >> 4;
  const int b = blockIdx.z, h = blockIdx.y;
  const int rBlk = blockIdx.x * 128;
  const int r0 = rBlk + wave * 16;
  const float gamma = 1.f - exp2f(-5.f - (float)h);
  const float lg = log2f(gamma);
  const size_t baseBH = (size_t)(b * S_) * HID_ + (size_t)h * HD_;

  // Q fragments for this 16-row strip (HD=128 -> 4 K-chunks of 32)
  v16h qf[4];
  {
    const h16_t* qrow = Q + baseBH + (size_t)(r0 + lane16) * HID_;
#pragma unroll
    for (int ec = 0; ec < 4; ++ec) {
      int kh = laneHi << 3;
      v8h lo = *(const v8h*)(qrow + ec*32 + kh);
      v8h hi = *(const v8h*)(qrow + ec*32 + 16 + kh);
      qf[ec] = cat8(lo, hi);
    }
  }
  const v8f vz = {0.f,0.f,0.f,0.f,0.f,0.f,0.f,0.f};
  v8f yacc[8];
#pragma unroll
  for (int nt = 0; nt < 8; ++nt) yacc[nt] = vz;

  const int nkb = (rBlk >> 5) + 4;          // key blocks of 32 up to rBlk+127
  for (int kb = 0; kb < nkb; ++kb) {
    const int kbase = kb << 5;
    // stage V^T chunk: 32 keys x 128 e -> sV[e][key]
#pragma unroll
    for (int i = 0; i < 2; ++i) {
      int c = tid + (i << 8);
      int kr = c >> 4;
      int ecol = (c & 15) << 3;
      v8h vv = *(const v8h*)(V + baseBH + (size_t)(kbase + kr) * HID_ + ecol);
#pragma unroll
      for (int j = 0; j < 8; ++j) sV[(ecol + j)*32 + kr] = vv[j];
    }
    __syncthreads();

    if (kbase <= r0 + 15) {                 // wave-uniform: causal skip
#pragma unroll
      for (int t = 0; t < 2; ++t) {
        const int kt = kbase + t * 16;
        v8f sc = vz;
        const h16_t* krow = Kh + baseBH + (size_t)(kt + lane16) * HID_;
#pragma unroll
        for (int ec = 0; ec < 4; ++ec) {
          int eoff = ec*32 + (laneHi << 3);
          v8h lo = *(const v8h*)(krow + eoff);
          v8h hi = *(const v8h*)(krow + eoff + 16);
          sc = wmma16(qf[ec], cat8(lo, hi), sc);
        }
        // decay * causal mask, emit f16 score tile to LDS (C->A relayout)
#pragma unroll
        for (int r = 0; r < 8; ++r) {
          int m = r0 + r + (laneHi << 3);
          int kc = kt + lane16;
          int diff = m - kc;
          float w = (diff >= 0) ? exp2f(lg * (float)diff) : 0.f;
          sS[wave][(r + (laneHi << 3))*32 + t*16 + lane16] = (h16_t)(sc[r] * w);
        }
      }
      // scores(16x32) @ V^T(32x128)
      v16h sf;
      {
        int kh = laneHi << 3;
        v8h lo = *(const v8h*)(&sS[wave][lane16*32 + kh]);
        v8h hi = *(const v8h*)(&sS[wave][lane16*32 + 16 + kh]);
        sf = cat8(lo, hi);
      }
#pragma unroll
      for (int nt = 0; nt < 8; ++nt) {
        int e = nt*16 + lane16;
        int ko = laneHi << 3;
        v8h lo = *(const v8h*)(&sV[e*32 + ko]);
        v8h hi = *(const v8h*)(&sV[e*32 + ko + 16]);
        yacc[nt] = wmma16(sf, cat8(lo, hi), yacc[nt]);
      }
    }
    __syncthreads();
  }

  // Fused group norm over HD per row (row's 128 ch are 8 accs x 16 lanes).
#pragma unroll
  for (int r = 0; r < 8; ++r) {
    float s1 = 0.f, s2 = 0.f;
#pragma unroll
    for (int nt = 0; nt < 8; ++nt) { float v = yacc[nt][r]; s1 += v; s2 += v*v; }
    for (int off = 1; off < 16; off <<= 1) {
      s1 += __shfl_xor(s1, off, 32);
      s2 += __shfl_xor(s2, off, 32);
    }
    float mean = s1 * (1.f / HD_);
    float var  = s2 * (1.f / HD_) - mean * mean;
    float inv  = rsqrtf(var + EPS_);
    int m = r0 + r + (laneHi << 3);
    h16_t* orow = Yout + (size_t)(b * S_ + m) * HID_ + (size_t)h * HD_;
#pragma unroll
    for (int nt = 0; nt < 8; ++nt) {
      int e = nt*16 + lane16;
      int cch = h * HD_ + e;
      orow[e] = (h16_t)((yacc[nt][r] - mean) * inv * gn_g[cch] + gn_b[cch]);
    }
  }
}

// ---- elementwise / prep kernels -------------------------------------------

__global__ void k_tables(float* __restrict__ tab) {  // [su|cu|sd|cd][S,64]
  int idx = blockIdx.x * blockDim.x + threadIdx.x;
  if (idx >= S_ * 64) return;
  int s = idx >> 6, i = idx & 63;
  float d = (float)HD_;
  float scale = (2.f * (float)i + 0.4f * d) / (1.4f * d);
  float sp = powf(scale, (float)s / 512.0f);
  float inv_freq = powf(10000.f, -(float)i / 64.f);
  float ang = (float)s * inv_freq;
  float sn = sinf(ang), cs = cosf(ang);
  tab[idx]            = sn * sp;
  tab[S_*64 + idx]    = cs * sp;
  tab[2*S_*64 + idx]  = sn / sp;
  tab[3*S_*64 + idx]  = cs / sp;
}

// weight f32 [K,N] (x Hn batches) -> f16 [Hn*N, K]
__global__ void k_wt(const float* __restrict__ in, h16_t* __restrict__ out,
                     int K, int N, size_t inStride)
{
  int k  = blockIdx.y * 32 + threadIdx.x;
  int n0 = blockIdx.x * 32 + threadIdx.y * 4;
  int hz = blockIdx.z;
  const float* ip = in + (size_t)hz * inStride;
  h16_t* op = out + (size_t)hz * (size_t)N * (size_t)K;
#pragma unroll
  for (int i = 0; i < 4; ++i) {
    int n = n0 + i;
    op[(size_t)n * K + k] = (h16_t)ip[(size_t)k * N + n];
  }
}

__global__ void k_pack_zc(const float* __restrict__ z, const float* __restrict__ c,
                          h16_t* __restrict__ out)
{
  int i = blockIdx.x * 256 + threadIdx.x;
  if (i >= M_ * HID_) return;
  int m = i >> 10, j = i & 1023;
  float v = (j < 512) ? z[(size_t)m * 512 + j] : c[(size_t)m * 512 + (j - 512)];
  out[i] = (h16_t)v;
}

__global__ __launch_bounds__(256)
void k_layernorm(const float* __restrict__ X, const float* __restrict__ g,
                 const float* __restrict__ bt, h16_t* __restrict__ out)
{
  __shared__ float red1[8], red2[8];
  int row = blockIdx.x, tid = threadIdx.x;
  const float* xr = X + (size_t)row * HID_;
  v4f v = *(const v4f*)(xr + tid * 4);
  float s1 = v[0]+v[1]+v[2]+v[3];
  float s2 = v[0]*v[0]+v[1]*v[1]+v[2]*v[2]+v[3]*v[3];
  for (int off = 16; off; off >>= 1) {
    s1 += __shfl_xor(s1, off, 32);
    s2 += __shfl_xor(s2, off, 32);
  }
  if ((tid & 31) == 0) { red1[tid >> 5] = s1; red2[tid >> 5] = s2; }
  __syncthreads();
  float t1 = 0.f, t2 = 0.f;
#pragma unroll
  for (int i = 0; i < 8; ++i) { t1 += red1[i]; t2 += red2[i]; }
  float mean = t1 / HID_, var = t2 / HID_ - mean * mean;
  float inv = rsqrtf(var + EPS_);
  h16_t* orow = out + (size_t)row * HID_;
#pragma unroll
  for (int j = 0; j < 4; ++j) {
    int cc = tid * 4 + j;
    orow[cc] = (h16_t)((v[j] - mean) * inv * g[cc] + bt[cc]);
  }
}

__global__ void k_xpos(h16_t* __restrict__ Q, h16_t* __restrict__ Kh,
                       const float* __restrict__ tab)
{
  int idx = blockIdx.x * blockDim.x + threadIdx.x;   // over M_*H_*64
  if (idx >= M_ * H_ * 64) return;
  int i   = idx & 63;
  int hh  = (idx >> 6) & 7;
  int row = idx >> 9;
  int s   = row & (S_ - 1);
  float ssu = tab[s*64 + i];
  float scu = tab[S_*64 + s*64 + i];
  float ssd = tab[2*S_*64 + s*64 + i];
  float scd = tab[3*S_*64 + s*64 + i];
  size_t p = (size_t)row * HID_ + (size_t)hh * HD_ + 2 * i;
  float q0 = (float)Q[p], q1 = (float)Q[p+1];
  Q[p]   = (h16_t)(q0 * scu - q1 * ssu);
  Q[p+1] = (h16_t)(q1 * scu + q0 * ssu);
  float k0 = (float)Kh[p], k1 = (float)Kh[p+1];
  Kh[p]   = (h16_t)(k0 * scd - k1 * ssd);
  Kh[p+1] = (h16_t)(k1 * scd + k0 * ssd);
}

__global__ void k_gate(const float* __restrict__ G, const h16_t* __restrict__ Y,
                       h16_t* __restrict__ out, int n)
{
  int i = blockIdx.x * blockDim.x + threadIdx.x;
  if (i >= n) return;
  float g = G[i];
  float sg = 1.f / (1.f + expf(-g));
  out[i] = (h16_t)(g * sg * (float)Y[i]);
}

__global__ void k_tofp16(const float* __restrict__ in, h16_t* __restrict__ out, int n)
{
  int i = blockIdx.x * blockDim.x + threadIdx.x;
  if (i >= n) return;
  out[i] = (h16_t)in[i];
}

// ---- host orchestration ----------------------------------------------------

extern "C" void kernel_launch(void* const* d_in, const int* in_sizes, int n_in,
                              void* d_out, int out_size, void* d_ws, size_t ws_size,
                              hipStream_t stream)
{
  (void)in_sizes; (void)n_in; (void)out_size; (void)ws_size;
  const float* z = (const float*)d_in[0];
  const float* c = (const float*)d_in[1];
  // params assumed flattened in jax pytree (sorted-key) order:
  // in_W, in_b, layers[0..3]{WG,WK,WO,WQ,WV,f1_W,f1_b,f2_W,f2_b,gn_b,gn_g,
  //                          ln1_b,ln1_g,ln2_b,ln2_g}, lv_W, lv_b, mu_W, mu_b
  const float* in_W = (const float*)d_in[2];
  const float* in_b = (const float*)d_in[3];
  struct Layer {
    const float *WG,*WK,*WO,*WQ,*WV,*f1W,*f1b,*f2W,*f2b,*gnb,*gng,
                *ln1b,*ln1g,*ln2b,*ln2g;
  } L[4];
  for (int l = 0; l < 4; ++l) {
    const float** p = (const float**)&L[l];
    for (int j = 0; j < 15; ++j) p[j] = (const float*)d_in[4 + l*15 + j];
  }
  const float* lv_W = (const float*)d_in[64];
  const float* lv_b = (const float*)d_in[65];
  const float* mu_W = (const float*)d_in[66];
  const float* mu_b = (const float*)d_in[67];

  char* ws = (char*)d_ws;
  size_t off = 0;
  auto alloc = [&](size_t bytes) -> void* {
    void* p = ws + off;
    off = (off + bytes + 255) & ~(size_t)255;
    return p;
  };

  h16_t* zc16  = (h16_t*)alloc((size_t)M_*HID_*2);
  float* x     = (float*)alloc((size_t)M_*HID_*4);
  h16_t* xn16  = (h16_t*)alloc((size_t)M_*HID_*2);
  h16_t* q16   = (h16_t*)alloc((size_t)M_*HID_*2);
  h16_t* k16   = (h16_t*)alloc((size_t)M_*HID_*2);
  h16_t* v16   = (h16_t*)alloc((size_t)M_*HID_*2);
  float* g32   = (float*)alloc((size_t)M_*HID_*4);
  h16_t* ygn16 = (h16_t*)alloc((size_t)M_*HID_*2);
  h16_t* gate16= (h16_t*)alloc((size_t)M_*HID_*2);
  float* y32   = (float*)alloc((size_t)M_*HID_*4);
  h16_t* hh16  = (h16_t*)alloc((size_t)M_*FFN_*2);
  h16_t* x16   = (h16_t*)alloc((size_t)M_*HID_*2);
  float* tab   = (float*)alloc((size_t)4*S_*64*4);
  h16_t* w_in  = (h16_t*)alloc((size_t)HID_*HID_*2);
  h16_t *wq[4], *wk[4], *wv[4], *wg[4], *wo[4], *wf1[4], *wf2[4];
  for (int l = 0; l < 4; ++l) {
    wq[l]  = (h16_t*)alloc((size_t)HID_*HID_*2);
    wk[l]  = (h16_t*)alloc((size_t)HID_*HID_*2);
    wv[l]  = (h16_t*)alloc((size_t)HID_*HID_*2);
    wg[l]  = (h16_t*)alloc((size_t)HID_*HID_*2);
    wo[l]  = (h16_t*)alloc((size_t)HID_*HID_*2);
    wf1[l] = (h16_t*)alloc((size_t)FFN_*HID_*2);
    wf2[l] = (h16_t*)alloc((size_t)HID_*FFN_*2);
  }
  h16_t* wmu = (h16_t*)alloc((size_t)OUT_*HID_*2);
  h16_t* wlv = (h16_t*)alloc((size_t)OUT_*HID_*2);

  const dim3 tb(32, 8);
  k_tables<<<(S_*64 + 255)/256, 256, 0, stream>>>(tab);
  k_wt<<<dim3(HID_/32, HID_/32, 1), tb, 0, stream>>>(in_W, w_in, HID_, HID_, 0);
  for (int l = 0; l < 4; ++l) {
    k_wt<<<dim3(HD_/32, HID_/32, H_), tb, 0, stream>>>(L[l].WQ, wq[l], HID_, HD_, (size_t)HID_*HD_);
    k_wt<<<dim3(HD_/32, HID_/32, H_), tb, 0, stream>>>(L[l].WK, wk[l], HID_, HD_, (size_t)HID_*HD_);
    k_wt<<<dim3(HD_/32, HID_/32, H_), tb, 0, stream>>>(L[l].WV, wv[l], HID_, HD_, (size_t)HID_*HD_);
    k_wt<<<dim3(HID_/32, HID_/32, 1), tb, 0, stream>>>(L[l].WG, wg[l], HID_, HID_, 0);
    k_wt<<<dim3(HID_/32, HID_/32, 1), tb, 0, stream>>>(L[l].WO, wo[l], HID_, HID_, 0);
    k_wt<<<dim3(FFN_/32, HID_/32, 1), tb, 0, stream>>>(L[l].f1W, wf1[l], HID_, FFN_, 0);
    k_wt<<<dim3(HID_/32, FFN_/32, 1), tb, 0, stream>>>(L[l].f2W, wf2[l], FFN_, HID_, 0);
  }
  k_wt<<<dim3(OUT_/32, HID_/32, 1), tb, 0, stream>>>(mu_W, wmu, HID_, OUT_, 0);
  k_wt<<<dim3(OUT_/32, HID_/32, 1), tb, 0, stream>>>(lv_W, wlv, HID_, OUT_, 0);

  const int nEl = M_ * HID_;
  k_pack_zc<<<(nEl + 255)/256, 256, 0, stream>>>(z, c, zc16);

  const dim3 gb(HID_/128, M_/128);     // (8, 32)
  k_gemm<1><<<gb, 256, 0, stream>>>(zc16, w_in, in_b, nullptr, x, nullptr, M_, HID_, HID_);

  for (int l = 0; l < 4; ++l) {
    k_layernorm<<<M_, 256, 0, stream>>>(x, L[l].ln1g, L[l].ln1b, xn16);
    k_gemm<0><<<gb, 256, 0, stream>>>(xn16, wq[l], nullptr, nullptr, nullptr, q16, M_, HID_, HID_);
    k_gemm<0><<<gb, 256, 0, stream>>>(xn16, wk[l], nullptr, nullptr, nullptr, k16, M_, HID_, HID_);
    k_gemm<0><<<gb, 256, 0, stream>>>(xn16, wv[l], nullptr, nullptr, nullptr, v16, M_, HID_, HID_);
    k_xpos<<<(M_*H_*64 + 255)/256, 256, 0, stream>>>(q16, k16, tab);
    k_retention<<<dim3(S_/128, H_, B_), 256, 0, stream>>>(q16, k16, v16, L[l].gng, L[l].gnb, ygn16);
    k_gemm<0><<<gb, 256, 0, stream>>>(xn16, wg[l], nullptr, nullptr, g32, nullptr, M_, HID_, HID_);
    k_gate<<<(nEl + 255)/256, 256, 0, stream>>>(g32, ygn16, gate16, nEl);
    k_gemm<3><<<gb, 256, 0, stream>>>(gate16, wo[l], nullptr, x, y32, nullptr, M_, HID_, HID_);
    k_layernorm<<<M_, 256, 0, stream>>>(y32, L[l].ln2g, L[l].ln2b, xn16);
    const dim3 gf1(FFN_/128, M_/128);
    k_gemm<2><<<gf1, 256, 0, stream>>>(xn16, wf1[l], L[l].f1b, nullptr, nullptr, hh16, M_, FFN_, HID_);
    k_gemm<4><<<gb, 256, 0, stream>>>(hh16, wf2[l], L[l].f2b, y32, x, nullptr, M_, HID_, FFN_);
  }

  k_tofp16<<<(nEl + 255)/256, 256, 0, stream>>>(x, x16, nEl);
  const dim3 go(OUT_/128, M_/128);
  float* out = (float*)d_out;
  k_gemm<1><<<go, 256, 0, stream>>>(x16, wmu, mu_b, nullptr, out, nullptr, M_, OUT_, HID_);
  k_gemm<5><<<go, 256, 0, stream>>>(x16, wlv, lv_b, nullptr, out + (size_t)M_*OUT_, nullptr, M_, OUT_, HID_);
}
